// VarianceNetwork_35502199669081
// MI455X (gfx1250) — matrix operand order repaired and checked by reading, not compiled, not run on hardware
//
#include <hip/hip_runtime.h>

typedef __attribute__((ext_vector_type(16))) _Float16 v16h;
typedef __attribute__((ext_vector_type(8)))  _Float16 v8h;
typedef __attribute__((ext_vector_type(4)))  _Float16 v4h;
typedef __attribute__((ext_vector_type(8)))  float    v8f;

#define BATCH   8
#define NPTS    4096
#define MROWS   (BATCH * NPTS)     // 32768
#define FDIM    512
#define KP      544                // padded K (17 * 32)
#define NT      (FDIM / 16)        // 32 n-tiles
#define ROWS    32                 // M-tile per workgroup (2 m-subtiles)
#define XSTRIDE 552                // LDS row stride (halves, mult of 8)

// ---------------------------------------------------------------------------
// Weight prep: f32 -> f16, repacked into the B-fragment order used here (contiguous k per lane half):
//   packed[((kt*NT + nt)*32 + lane)*16 + j] = W[kt*32 + (lane&16?16:0) + j][nt*16 + (lane&15)]
// so a B fragment is one contiguous 32-byte load per lane.
// ---------------------------------------------------------------------------
__global__ __launch_bounds__(256)
void prep_weights(const float* __restrict__ w1, int K1,
                  const float* __restrict__ wr,
                  _Float16* __restrict__ w1p,
                  _Float16* __restrict__ wrp)
{
    int i = blockIdx.x * blockDim.x + threadIdx.x;
    if (i < KP * FDIM) {                       // (KP/32)*NT*32*16
        int j    = i & 15;
        int lane = (i >> 4) & 31;
        int tile = i >> 9;                     // kt*NT + nt
        int nt   = tile & (NT - 1);
        int kt   = tile >> 5;
        int k    = kt * 32 + ((lane & 16) ? 16 : 0) + j;
        int n    = nt * 16 + (lane & 15);
        w1p[i] = (_Float16)(k < K1 ? w1[k * FDIM + n] : 0.0f);
    }
    if (i < FDIM * FDIM) {
        int j    = i & 15;
        int lane = (i >> 4) & 31;
        int tile = i >> 9;
        int nt   = tile & (NT - 1);
        int kt   = tile >> 5;
        int k    = kt * 32 + ((lane & 16) ? 16 : 0) + j;
        int n    = nt * 16 + (lane & 15);
        wrp[i] = (_Float16)wr[k * FDIM + n];
    }
}

// ---------------------------------------------------------------------------
// Local covariance feature: top-2 neighbors (self + nearest), outer product.
// points layout in memory: [B, 3, N]
// ---------------------------------------------------------------------------
__global__ __launch_bounds__(256)
void local_cov_kernel(const float* __restrict__ rp, float* __restrict__ local)
{
    __shared__ float px[NPTS];
    __shared__ float py[NPTS];
    __shared__ float pz[NPTS];

    const int b = blockIdx.y;
    const float* base = rp + (size_t)b * 3 * NPTS;
    for (int i = threadIdx.x; i < NPTS; i += blockDim.x) {
        px[i] = base[i];
        py[i] = base[NPTS + i];
        pz[i] = base[2 * NPTS + i];
    }
    __syncthreads();

    const int n = blockIdx.x * blockDim.x + threadIdx.x;
    const float qx = px[n], qy = py[n], qz = pz[n];

    // top-2 of pd = -squared distance; strict '>' keeps lowest index on ties
    float best0 = -3.4e38f, best1 = -3.4e38f;
    int   i0 = 0, i1 = 0;
    for (int j = 0; j < NPTS; ++j) {
        float dx = qx - px[j], dy = qy - py[j], dz = qz - pz[j];
        float pd = -(dx * dx + dy * dy + dz * dz);
        if (pd > best0) {
            best1 = best0; i1 = i0;
            best0 = pd;    i0 = j;
        } else if (pd > best1) {
            best1 = pd;    i1 = j;
        }
    }

    const float ax = px[i0], ay = py[i0], az = pz[i0];
    const float bx = px[i1], by = py[i1], bz = pz[i1];

    float* o = local + ((size_t)b * NPTS + n) * 12;
    o[0]  = qx;      o[1]  = qy;      o[2]  = qz;
    o[3]  = ax * bx; o[4]  = ax * by; o[5]  = ax * bz;
    o[6]  = ay * bx; o[7]  = ay * by; o[8]  = ay * bz;
    o[9]  = az * bx; o[10] = az * by; o[11] = az * bz;
}

// ---------------------------------------------------------------------------
// A fragment (16x32 f16, k = 16*hi + s per lane, as in packed B): two 16-byte contiguous runs per lane.
// ---------------------------------------------------------------------------
__device__ __forceinline__
v16h load_a_frag(const _Float16* rowbase, int kb)
{
    v8h lo = *(const v8h*)(rowbase + kb);
    v8h hh = *(const v8h*)(rowbase + kb + 8);
    return __builtin_shufflevector(lo, hh, 0, 1, 2, 3, 4, 5, 6, 7,
                                            8, 9, 10, 11, 12, 13, 14, 15);
}

// ---------------------------------------------------------------------------
// Fused stage: h1 = relu(x@W1+b1); h2 = relu(h1 + h1@Wr+br); out = h2@w2+b2
// One WG = 32 rows x 512 cols, single LDS buffer reused x -> h1 -> h2.
// 8 waves; each wave: 2 m-subtiles x 4 n-tiles = 8 WMMAs per k-step,
// reusing each B fragment across both m-subtiles (2x L2 traffic reduction).
// ---------------------------------------------------------------------------
__global__ __launch_bounds__(256)
void stage_kernel(const float* __restrict__ latent,
                  const float* __restrict__ extra, int E,
                  const _Float16* __restrict__ w1p,
                  const float* __restrict__ b1,
                  const _Float16* __restrict__ wrp,
                  const float* __restrict__ br,
                  const float* __restrict__ w2,
                  const float* __restrict__ b2,
                  const float* __restrict__ prev, int addprev,
                  float* __restrict__ outstd)
{
    __shared__ __align__(16) _Float16 xs[ROWS][XSTRIDE];   // x -> h1 -> h2

    const int tid  = threadIdx.x;
    const int row0 = blockIdx.x * ROWS;
    const int wave = tid >> 5;
    const int lane = tid & 31;
    const int nlane = lane & 15;
    const int hi    = (lane & 16) ? 1 : 0;

    // ---- stage input features -> LDS f16 (vectorized), zero-padded to KP ----
    for (int i = tid; i < ROWS * (FDIM / 4); i += 256) {
        int r  = i >> 7;                 // FDIM/4 = 128
        int c4 = i & 127;
        const float4 v =
            ((const float4*)(latent + (size_t)(row0 + r) * FDIM))[c4];
        v4h h = {(_Float16)v.x, (_Float16)v.y, (_Float16)v.z, (_Float16)v.w};
        *(v4h*)&xs[r][c4 * 4] = h;
    }
    for (int i = tid; i < ROWS * (KP - FDIM); i += 256) {
        int r = i >> 5;                  // KP-FDIM = 32
        int c = FDIM + (i & 31);
        float v = (c < FDIM + E) ? extra[(size_t)(row0 + r) * E + (c - FDIM)]
                                 : 0.0f;
        xs[r][c] = (_Float16)v;
    }
    __syncthreads();

    const v16h* w1v = (const v16h*)w1p;
    const v16h* wrv = (const v16h*)wrp;

    // ================= GEMM 1: h1 = relu(x @ W1 + b1) =================
    v8f acc[2][4] = {};
    for (int kt = 0; kt < KP / 32; ++kt) {
        const int kb = kt * 32 + (hi ? 16 : 0);
        v16h afr[2];
        #pragma unroll
        for (int mt = 0; mt < 2; ++mt)
            afr[mt] = load_a_frag(&xs[mt * 16 + nlane][0], kb);
        #pragma unroll
        for (int t = 0; t < 4; ++t) {
            const int ntile = wave * 4 + t;
            const v16h* bp = w1v + ((size_t)(kt * NT + ntile) * 32 + lane);
            if (kt + 1 < KP / 32)
                __builtin_prefetch(bp + (size_t)NT * 32, 0, 1); // next k-step
            const v16h bfr = *bp;
            #pragma unroll
            for (int mt = 0; mt < 2; ++mt)
                acc[mt][t] = __builtin_amdgcn_wmma_f32_16x16x32_f16(
                    false, afr[mt], false, bfr, (short)0, acc[mt][t],
                    false, false);
        }
    }
    __syncthreads();   // all A reads of x done before h1 overwrites xs
    #pragma unroll
    for (int t = 0; t < 4; ++t) {
        const int col = wave * 64 + t * 16 + nlane;
        const float bias = b1[col];
        #pragma unroll
        for (int mt = 0; mt < 2; ++mt) {
            #pragma unroll
            for (int r = 0; r < 8; ++r) {
                float v = acc[mt][t][r] + bias;
                xs[mt * 16 + hi * 8 + r][col] = (_Float16)(v > 0.0f ? v : 0.0f);
            }
        }
    }
    __syncthreads();

    // ============ GEMM 2: h2 = relu(h1 + h1 @ Wr + br) ============
    v8f acc2[2][4] = {};
    for (int kt = 0; kt < FDIM / 32; ++kt) {
        const int kb = kt * 32 + (hi ? 16 : 0);
        v16h afr[2];
        #pragma unroll
        for (int mt = 0; mt < 2; ++mt)
            afr[mt] = load_a_frag(&xs[mt * 16 + nlane][0], kb);
        #pragma unroll
        for (int t = 0; t < 4; ++t) {
            const int ntile = wave * 4 + t;
            const v16h* bp = wrv + ((size_t)(kt * NT + ntile) * 32 + lane);
            if (kt + 1 < FDIM / 32)
                __builtin_prefetch(bp + (size_t)NT * 32, 0, 1);
            const v16h bfr = *bp;
            #pragma unroll
            for (int mt = 0; mt < 2; ++mt)
                acc2[mt][t] = __builtin_amdgcn_wmma_f32_16x16x32_f16(
                    false, afr[mt], false, bfr, (short)0, acc2[mt][t],
                    false, false);
        }
    }
    __syncthreads();   // all A reads of h1 done before h2 overwrites xs
    #pragma unroll
    for (int t = 0; t < 4; ++t) {
        const int col = wave * 64 + t * 16 + nlane;
        const float bias = br[col];
        #pragma unroll
        for (int mt = 0; mt < 2; ++mt) {
            #pragma unroll
            for (int r = 0; r < 8; ++r) {
                const int m = mt * 16 + hi * 8 + r;
                float v = (float)xs[m][col] + acc2[mt][t][r] + bias;
                xs[m][col] = (_Float16)(v > 0.0f ? v : 0.0f);  // h2 in place
            }
        }
    }
    __syncthreads();

    // ================= GEMV: out = h2 @ w2 + b2 (+prev) =================
    {
        const int r    = tid >> 3;   // 32 rows, 8 threads per row
        const int lsub = tid & 7;
        float s = 0.0f;
        for (int c = lsub; c < FDIM; c += 8)
            s += (float)xs[r][c] * w2[c];
        s += __shfl_xor(s, 4, 32);
        s += __shfl_xor(s, 2, 32);
        s += __shfl_xor(s, 1, 32);
        if (lsub == 0) {
            const int row = row0 + r;
            float base = addprev ? prev[row] : 0.0f;
            outstd[row] = base + s + b2[0];
        }
    }
}

// ---------------------------------------------------------------------------
extern "C" void kernel_launch(void* const* d_in, const int* in_sizes, int n_in,
                              void* d_out, int out_size, void* d_ws, size_t ws_size,
                              hipStream_t stream)
{
    const float* latent = (const float*)d_in[0];
    const float* rpts   = (const float*)d_in[1];
    // per stage: w1, b1, wr, br, w2, b2
    const float* w[3][6];
    for (int s = 0; s < 3; ++s)
        for (int j = 0; j < 6; ++j)
            w[s][j] = (const float*)d_in[2 + s * 6 + j];

    char* ws = (char*)d_ws;
    float* local = (float*)ws;  ws += (size_t)MROWS * 12 * sizeof(float);
    float* std1  = (float*)ws;  ws += (size_t)MROWS * sizeof(float);
    float* std2  = (float*)ws;  ws += (size_t)MROWS * sizeof(float);
    _Float16* wp[3][2];
    for (int s = 0; s < 3; ++s) {
        wp[s][0] = (_Float16*)ws; ws += (size_t)KP * FDIM * sizeof(_Float16);
        wp[s][1] = (_Float16*)ws; ws += (size_t)FDIM * FDIM * sizeof(_Float16);
    }

    const int K1s[3] = {524, 513, 513};
    for (int s = 0; s < 3; ++s)
        prep_weights<<<(KP * FDIM + 255) / 256, 256, 0, stream>>>(
            w[s][0], K1s[s], w[s][2], wp[s][0], wp[s][1]);

    local_cov_kernel<<<dim3(NPTS / 256, BATCH), 256, 0, stream>>>(rpts, local);

    // stage 1: extra = local cov (12), no previous std
    stage_kernel<<<MROWS / ROWS, 256, 0, stream>>>(
        latent, local, 12,
        wp[0][0], w[0][1], wp[0][1], w[0][3], w[0][4], w[0][5],
        std1, 0, std1);
    // stage 2: extra = std1 (1), accumulate onto std1
    stage_kernel<<<MROWS / ROWS, 256, 0, stream>>>(
        latent, std1, 1,
        wp[1][0], w[1][1], wp[1][1], w[1][3], w[1][4], w[1][5],
        std1, 1, std2);
    // stage 3: extra = std2 (1), accumulate onto std2 -> final output
    stage_kernel<<<MROWS / ROWS, 256, 0, stream>>>(
        latent, std2, 1,
        wp[2][0], w[2][1], wp[2][1], w[2][3], w[2][4], w[2][5],
        std2, 1, (float*)d_out);
}
